// BasicPropertiesRCNNHead_20057497272473
// MI455X (gfx1250) — matrix operand order repaired and checked by reading, not compiled, not run on hardware
//
#include <hip/hip_runtime.h>
#include <hip/hip_bf16.h>

// ---------------------------------------------------------------------------
// Feature probes
// ---------------------------------------------------------------------------
#if defined(__has_builtin)
#  if __has_builtin(__builtin_amdgcn_global_load_async_to_lds_b128) && \
      __has_builtin(__builtin_amdgcn_s_wait_asynccnt)
#    define HAVE_ASYNC_LDS 1
#  endif
#  if __has_builtin(__builtin_amdgcn_cvt_pk_bf16_f32)
#    define HAVE_PK_BF16 1
#  endif
#endif
#ifndef HAVE_ASYNC_LDS
#  define HAVE_ASYNC_LDS 0
#endif
#ifndef HAVE_PK_BF16
#  define HAVE_PK_BF16 0
#endif

// ---------------------------------------------------------------------------
// Types
// ---------------------------------------------------------------------------
typedef __attribute__((ext_vector_type(16))) __bf16        v16bf;
typedef __attribute__((ext_vector_type(2)))  __bf16        v2bf;
typedef __attribute__((ext_vector_type(8)))  float         v8f;
typedef __attribute__((ext_vector_type(4)))  float         f4;
typedef __attribute__((ext_vector_type(4)))  unsigned int  u4;
typedef __attribute__((ext_vector_type(2)))  unsigned int  u2;
typedef __attribute__((ext_vector_type(4)))  int           v4i;

union Frag { u4 q[2]; v16bf v; };

// round-to-nearest (adds 0x8000; differs from RNE only on exact ties)
__device__ __forceinline__ unsigned int f2bf(float f) {
  return (__float_as_uint(f) + 0x8000u) >> 16;
}

// pack two f32 -> bf16x2: 2x v_add + 1x v_perm (or 1x v_cvt_pk if available)
__device__ __forceinline__ unsigned int pack2(float a, float b) {
#if HAVE_PK_BF16
  union { v2bf v; unsigned int u; } c;
  c.v = __builtin_amdgcn_cvt_pk_bf16_f32(a, b);
  return c.u;
#else
  unsigned int ra = __float_as_uint(a) + 0x8000u;
  unsigned int rb = __float_as_uint(b) + 0x8000u;
  // dst bytes: [0,1] = ra bytes [2,3] (sel 2,3) ; [2,3] = rb bytes [2,3] (sel 6,7)
  return __builtin_amdgcn_perm(rb, ra, 0x07060302u);
#endif
}

// ---------------------------------------------------------------------------
// Weight pack kernels: f32 [C_out, K] row-major  ->  bf16 WMMA-tiled
// layout: dst[((k>>5)*288 + c)*32 + (k&31)]  (K-tile major, then column,
// then 32 contiguous k) so a BK=32 B-tile is one contiguous 18KB block and
// each lane's B fragment is two 16-byte runs in LDS.
// ---------------------------------------------------------------------------
__global__ __launch_bounds__(256) void pack_fc1_kernel(const float* __restrict__ w,
                                                       unsigned short* __restrict__ dst) {
  int e2 = blockIdx.x * 256 + threadIdx.x;        // pair index (2 elems along k)
  if (e2 >= (288 * 12544) / 2) return;
  int e = e2 * 2;
  int c = e / 12544;
  int d = e - c * 12544;                          // d, d+1 in same 32-run
  unsigned int p = pack2(w[e], w[e + 1]);
  unsigned int di = ((((unsigned)(d >> 5)) * 288u + (unsigned)c) << 5) | (unsigned)(d & 31);
  *(unsigned int*)&dst[di] = p;                   // di even: 4B-aligned store
}

__global__ __launch_bounds__(256) void pack_fc2_kernel(const float* __restrict__ w,
                                                       unsigned short* __restrict__ dst) {
  int e2 = blockIdx.x * 256 + threadIdx.x;
  if (e2 >= (288 * 288) / 2) return;
  int e = e2 * 2;
  int n = e / 288;              // output class (B's N dim)
  int k = e - n * 288;          // input feature (B's K dim); k, k+1 in same 32-run
  unsigned int p = pack2(w[e], w[e + 1]);
  unsigned int di = ((((unsigned)(k >> 5)) * 288u + (unsigned)n) << 5) | (unsigned)(k & 31);
  *(unsigned int*)&dst[di] = p;
}

// ---------------------------------------------------------------------------
// Async B-tile copy: one contiguous 18KB block -> LDS via async-to-LDS.
// Waves 0-3 each issue exactly 9 b128 instructions (uniform per-wave ASYNCcnt).
// ---------------------------------------------------------------------------
#if HAVE_ASYNC_LDS
__device__ __forceinline__ void issue_async_B(const unsigned short* Bt_tile,
                                              unsigned int* BbufBase,
                                              int wid, int lane) {
  if (wid < 4) {
    const u4* src = (const u4*)Bt_tile;
    u4* dst = (u4*)BbufBase;
#pragma unroll
    for (int i = 0; i < 9; ++i) {
      const int idx = wid * 288 + i * 32 + lane;   // 4*288 = 1152 uint4 = 18KB
      __builtin_amdgcn_global_load_async_to_lds_b128(
          (__attribute__((address_space(1))) v4i*)(src + idx),
          (__attribute__((address_space(3))) v4i*)(dst + idx),
          0, 0);
    }
  }
}
#endif

// ---------------------------------------------------------------------------
// bf16 WMMA GEMM:  Out[M,288] = act(A[M,K] x Bt + bias)
//   AMODE 0: A is f32 (x), converted to bf16 while staging into LDS
//   AMODE 1: A is bf16 (h)
//   OMODE 0: relu -> bf16 store       (GEMM1 -> h)
//   OMODE 1: relu -> sigmoid -> f32   (GEMM2 -> prob)
// Block: 256 threads (8 waves). Tile: BM=64 x BN=288 (full N => A read once),
// BK=32. B tile: async-to-LDS double buffer; A tile: register double buffer.
// Wave map: (wid&3) -> M strip of 16, (wid>>2) -> 144-col half (9 N-tiles).
// ---------------------------------------------------------------------------
template <int AMODE, int OMODE>
__global__ __launch_bounds__(256) void gemm_wmma_kernel(
    const void* __restrict__ Aptr, const unsigned short* __restrict__ Bt,
    const float* __restrict__ bias, void* __restrict__ Out, int K) {
  __shared__ __align__(16) unsigned int Abuf[64 * 16];          //  4 KB: [row][32 bf16]
#if HAVE_ASYNC_LDS
  __shared__ __align__(16) unsigned int Bbuf[2][288 * 16];      // 2x18 KB: [col][32 bf16]
#else
  __shared__ __align__(16) unsigned int Bbuf[1][288 * 16];      // 18 KB
#endif

  const int tid  = threadIdx.x;
  const int wid  = tid >> 5;
  const int lane = tid & 31;
  const int half = lane >> 4;           // 0: lanes 0-15, 1: lanes 16-31
  const int l15  = lane & 15;
  const int m0   = blockIdx.x * 64;
  const int colW = (wid >> 2) * 144;    // wave's N base (0 or 144)
  const int rowA = (wid & 3) * 16 + l15;
  const int KT   = K >> 5;

  v8f acc[9];
#pragma unroll
  for (int j = 0; j < 9; ++j)
#pragma unroll
    for (int r = 0; r < 8; ++r) acc[j][r] = 0.0f;

  // A staging maps (AMODE 0: 512 float4 / tile, 2 per thread)
  const int aRow0 = tid >> 3, aQ = tid & 7;
  const int aRow1 = (tid + 256) >> 3;
  // AMODE 1: 256 uint4 / tile, 1 per thread
  const int hRow = tid >> 2, hQ = tid & 3;

  f4 apf0, apf1;   // AMODE 0 prefetch regs
  u4 apfb;         // AMODE 1 prefetch reg
#if !HAVE_ASYNC_LDS
  u4 bpf[5];       // B prefetch regs (1152 uint4 / tile)
#endif

  // ---- prefetch K-tile 0
  if (AMODE == 0) {
    const float* A = (const float*)Aptr;
    apf0 = *(const f4*)(A + (size_t)(m0 + aRow0) * K + aQ * 4);
    apf1 = *(const f4*)(A + (size_t)(m0 + aRow1) * K + aQ * 4);
  } else {
    const unsigned short* A = (const unsigned short*)Aptr;
    apfb = *(const u4*)(A + (size_t)(m0 + hRow) * K + hQ * 8);
  }
#if HAVE_ASYNC_LDS
  issue_async_B(Bt, Bbuf[0], wid, lane);
#else
  {
    const u4* Bs = (const u4*)Bt;
#pragma unroll
    for (int j = 0; j < 4; ++j) bpf[j] = Bs[tid + j * 256];
    if (tid < 128) bpf[4] = Bs[tid + 1024];
  }
#endif

  for (int kt = 0; kt < KT; ++kt) {
    __syncthreads();     // all waves done with prior compute; LDS write-safe
    // ---- commit staged A registers to LDS
    if (AMODE == 0) {
      u2 p0, p1;
      p0[0] = pack2(apf0[0], apf0[1]); p0[1] = pack2(apf0[2], apf0[3]);
      p1[0] = pack2(apf1[0], apf1[1]); p1[1] = pack2(apf1[2], apf1[3]);
      *(u2*)&Abuf[aRow0 * 16 + aQ * 2] = p0;
      *(u2*)&Abuf[aRow1 * 16 + aQ * 2] = p1;
    } else {
      *(u4*)&Abuf[hRow * 16 + hQ * 4] = apfb;
    }
#if !HAVE_ASYNC_LDS
    {
      u4* Bl = (u4*)Bbuf[0];
#pragma unroll
      for (int j = 0; j < 4; ++j) Bl[tid + j * 256] = bpf[j];
      if (tid < 128) Bl[tid + 1024] = bpf[4];
    }
#endif
    // ---- issue next-tile prefetches (overlap with compute)
    if (kt + 1 < KT) {
      const int kn = (kt + 1) * 32;
      if (AMODE == 0) {
        const float* A = (const float*)Aptr;
        apf0 = *(const f4*)(A + (size_t)(m0 + aRow0) * K + kn + aQ * 4);
        apf1 = *(const f4*)(A + (size_t)(m0 + aRow1) * K + kn + aQ * 4);
      } else {
        const unsigned short* A = (const unsigned short*)Aptr;
        apfb = *(const u4*)(A + (size_t)(m0 + hRow) * K + kn + hQ * 8);
      }
#if HAVE_ASYNC_LDS
      issue_async_B(Bt + (size_t)(kt + 1) * 9216, Bbuf[(kt + 1) & 1], wid, lane);
#else
      const u4* Bs = (const u4*)(Bt + (size_t)(kt + 1) * 9216);
#pragma unroll
      for (int j = 0; j < 4; ++j) bpf[j] = Bs[tid + j * 256];
      if (tid < 128) bpf[4] = Bs[tid + 1024];
#endif
    }
#if HAVE_ASYNC_LDS
    // tile kt has landed when <= 9 async ops remain (the tile kt+1 issue)
    if (kt + 1 < KT) __builtin_amdgcn_s_wait_asynccnt(9);
    else             __builtin_amdgcn_s_wait_asynccnt(0);
#endif
    __syncthreads();     // staging (ds stores + async completions) visible

    // ---- compute: 9 WMMAs per wave on this K-slab
    const unsigned int* Bc = Bbuf[HAVE_ASYNC_LDS ? (kt & 1) : 0];
    Frag af;
    // A 16x32 bf16 layout: lanes 0-15 hold K0-7 (b0-15) + K16-23 (b32-47),
    // lanes 16-31 hold K8-15 (b16-31) + K24-31 (b48-63)
    af.q[0] = *(const u4*)&Abuf[rowA * 16 + half * 4];
    af.q[1] = *(const u4*)&Abuf[rowA * 16 + 8 + half * 4];
#pragma unroll
    for (int j = 0; j < 9; ++j) {
      Frag bfg;
      const int col = colW + j * 16 + l15;
      // B 32x16 bf16 layout: lane's column, K8h..8h+7 then K16+8h..16+8h+7 (matches A)
      bfg.q[0] = *(const u4*)&Bc[col * 16 + half * 4];
      bfg.q[1] = *(const u4*)&Bc[col * 16 + 8 + half * 4];
      acc[j] = __builtin_amdgcn_wmma_f32_16x16x32_bf16(
          false, af.v, false, bfg.v, (short)0, acc[j], false, false);
    }
  }

  // ---- epilogue: C/D layout: lanes 0-15 -> M=r, lanes 16-31 -> M=r+8, N=l15
  const int gRow0 = m0 + (wid & 3) * 16 + (half ? 8 : 0);
#pragma unroll
  for (int j = 0; j < 9; ++j) {
    const int col = colW + j * 16 + l15;
    const float bs = bias[col];
#pragma unroll
    for (int r = 0; r < 8; ++r) {
      float v = acc[j][r] + bs;
      v = v > 0.0f ? v : 0.0f;                       // relu
      const size_t o = (size_t)(gRow0 + r) * 288 + col;
      if (OMODE == 0) {
        ((unsigned short*)Out)[o] = (unsigned short)f2bf(v);
      } else {
        ((float*)Out)[o] = 1.0f / (1.0f + __expf(-v));   // sigmoid
      }
    }
  }
}

// ---------------------------------------------------------------------------
// Thresholded top-3 per row (reversed order, invalid -> index 0)
// ---------------------------------------------------------------------------
__global__ __launch_bounds__(256) void top3_kernel(const float* __restrict__ prob,
                                                   int* __restrict__ idxOut) {
  int row = blockIdx.x * 256 + threadIdx.x;
  if (row >= 16384) return;
  const float* p = prob + (size_t)row * 288;
  float v0 = -1.0f, v1 = -1.0f, v2 = -1.0f;
  int i0 = 0, i1 = 0, i2 = 0;
  for (int c = 0; c < 288; ++c) {
    float v = p[c];
    if (v > 0.5f) {           // strict > keeps first-occurrence order on ties
      if (v > v0)      { v2 = v1; i2 = i1; v1 = v0; i1 = i0; v0 = v; i0 = c; }
      else if (v > v1) { v2 = v1; i2 = i1; v1 = v;  i1 = c; }
      else if (v > v2) { v2 = v;  i2 = c; }
    }
  }
  if (!(v0 > 0.5f)) i0 = 0;
  if (!(v1 > 0.5f)) i1 = 0;
  if (!(v2 > 0.5f)) i2 = 0;
  idxOut[row * 3 + 0] = i2;   // reversed: [:, ::-1]
  idxOut[row * 3 + 1] = i1;
  idxOut[row * 3 + 2] = i0;
}

// ---------------------------------------------------------------------------
// Launcher
// ---------------------------------------------------------------------------
extern "C" void kernel_launch(void* const* d_in, const int* in_sizes, int n_in,
                              void* d_out, int out_size, void* d_ws, size_t ws_size,
                              hipStream_t stream) {
  (void)in_sizes; (void)n_in; (void)out_size; (void)ws_size;

  const float* x     = (const float*)d_in[0];   // [16384, 256, 7, 7] = [16384, 12544]
  const float* fc1_w = (const float*)d_in[1];   // [288, 12544]
  const float* fc1_b = (const float*)d_in[2];   // [288]
  const float* fc2_w = (const float*)d_in[3];   // [288, 288]
  const float* fc2_b = (const float*)d_in[4];   // [288]

  // workspace layout (bf16): packed W1 | packed W2 | h
  unsigned short* fc1wT = (unsigned short*)d_ws;          // 3,612,672 elems
  unsigned short* fc2wT = fc1wT + 3612672;                //    82,944 elems
  unsigned short* hbuf  = fc2wT + 82944;                  // 4,718,592 elems

  float* prob   = (float*)d_out;                          // [16384, 288] f32
  int*   idxOut = (int*)(prob + (size_t)16384 * 288);     // [16384, 3] i32

  pack_fc1_kernel<<<(288 * 12544 / 2 + 255) / 256, 256, 0, stream>>>(fc1_w, fc1wT);
  pack_fc2_kernel<<<(288 * 288 / 2 + 255) / 256, 256, 0, stream>>>(fc2_w, fc2wT);

  // GEMM1: x(f32->bf16) x W1^T -> h (relu, bf16). 16384/64 = 256 blocks.
  gemm_wmma_kernel<0, 0><<<256, 256, 0, stream>>>(x, fc1wT, fc1_b, hbuf, 12544);
  // GEMM2: h(bf16) x W2^T -> prob (relu+sigmoid, f32).
  gemm_wmma_kernel<1, 1><<<256, 256, 0, stream>>>(hbuf, fc2wT, fc2_b, prob, 288);

  top3_kernel<<<64, 256, 0, stream>>>(prob, idxOut);
}
